// Conv2d_32091995636118
// MI455X (gfx1250) — matrix operand order repaired and checked by reading, not compiled, not run on hardware
//
#include <hip/hip_runtime.h>
#include <stdint.h>

typedef __attribute__((ext_vector_type(16))) __bf16 v16bf;
typedef __attribute__((ext_vector_type(8)))  float  v8f;
typedef __attribute__((ext_vector_type(4)))  unsigned int u32x4;
typedef __attribute__((ext_vector_type(8)))  int  i32x8;
typedef __attribute__((ext_vector_type(4)))  int  i32x4;

#define CIN_  64
#define COUT_ 64
#define H_    112
#define W_    112
#define NB_   16

// ---- LDS layout ---------------------------------------------------------
// sIn : [130 cols][stride 264 u16] : col*264 + r*64 + c  (4 halo rows) 68,640 B
// sW  : [9 taps][64 cout][stride 72 u16] : (t*64+co)*72 + ch          82,944 B
// sBias: 64 f32                                                           256 B
#define SIN_U16   34320              // 130*264
#define SW_U16    41472              // 9*64*72
#define SW_LDS_BYTE_OFF  68640       // dynamic LDS starts at 0 (no static LDS)
#define SMEM_BYTES (2*(SIN_U16 + SW_U16) + 64*4)   // 151,840

__device__ __forceinline__ uint16_t f2bf(float f) {
  union { float f; unsigned u; } v; v.f = f;
  unsigned r = 0x7FFFu + ((v.u >> 16) & 1u);
  return (uint16_t)((v.u + r) >> 16);
}

// Repack weights (COUT, CIN, 3, 3) f32 -> bf16 [tap][cout][cin] (dense)
__global__ void prep_weights(const float* __restrict__ w, uint16_t* __restrict__ wq) {
  int gid = blockIdx.x * blockDim.x + threadIdx.x;
  if (gid >= 9 * 64 * 64) return;
  int c    = gid & 63;
  int cout = (gid >> 6) & 63;
  int t    = gid >> 12;
  int r = t / 3, s = t - 3 * r;
  wq[gid] = f2bf(w[((cout * CIN_ + c) * 3 + r) * 3 + s]);
}

__global__ void __launch_bounds__(256, 2)
conv_wmma(const float* __restrict__ x, const uint16_t* __restrict__ wq,
          const float* __restrict__ bias, float* __restrict__ out) {
  extern __shared__ __align__(16) uint16_t smem[];
  uint16_t* sIn   = smem;
  uint16_t* sW    = smem + SIN_U16;
  float*    sBias = (float*)(smem + SIN_U16 + SW_U16);

  const int tid = threadIdx.x;
  const int n  = blockIdx.x / (H_ / 2);
  const int h0 = (blockIdx.x % (H_ / 2)) * 2;      // block covers rows h0, h0+1

  if (tid < 64) sBias[tid] = bias[tid];

  // ---- weights: one TDM descriptor, 9216 x 8B units, LDS-padded 128B->+16B
  if (tid < 32) {
    const uint64_t ga = (uint64_t)(uintptr_t)wq;
    u32x4 g0;
    g0[0] = 1u;                                     // count=1, load, no gather
    g0[1] = (unsigned)SW_LDS_BYTE_OFF;              // lds_addr
    g0[2] = (unsigned)(ga & 0xFFFFFFFFu);           // global_addr[31:0]
    g0[3] = (unsigned)((ga >> 32) & 0x01FFFFFFu) | (2u << 30); // addr[56:32], type=2
    i32x8 g1;
    g1[0] = (int)((3u << 16)      // data_size = 8B
                | (1u << 20)      // pad_enable
                | (4u << 22)      // pad_interval: every 32 dwords (one cout row)
                | (3u << 25));    // pad_amount: 4 dwords (16B) -> 144B stride
    g1[1] = (int)((9216u & 0xFFFFu) << 16);         // tensor_dim0 lo16 (abar=0)
    g1[2] = (int)((9216u >> 16) | (1u << 16));      // tensor_dim0 hi16 | tensor_dim1=1
    g1[3] = (int)((9216u & 0xFFFFu) << 16);         // tile_dim0 = 9216
    g1[4] = 1;                                      // tile_dim1 = 1, tile_dim2 = 0
    g1[5] = 9216;                                   // tensor_dim0_stride lo32
    g1[6] = 0;                                      // stride0 hi | stride1 lo
    g1[7] = 0;                                      // stride1 hi
    i32x4 g2 = {0, 0, 0, 0};
    i32x4 g3 = {0, 0, 0, 0};
    i32x8 g4 = {0, 0, 0, 0, 0, 0, 0, 0};
    __builtin_amdgcn_tensor_load_to_lds(g0, g1, g2, g3, g4, 0);
    __builtin_amdgcn_s_wait_tensorcnt(0);
  }

  // ---- stage input halo (4 rows x 64 ch x 130 cols), f32 -> bf16, zero-pad
  for (int i = tid; i < 4 * 64 * 130; i += 256) {
    int col = i % 130;
    int rc  = i / 130;
    int r   = rc >> 6;           // 0..3
    int c   = rc & 63;
    int wi  = col - 1;
    int hi  = h0 + r - 1;
    float v = 0.f;
    if ((unsigned)wi < (unsigned)W_ && (unsigned)hi < (unsigned)H_)
      v = x[(((size_t)n * CIN_ + c) * H_ + hi) * W_ + wi];
    sIn[col * 264 + r * 64 + c] = f2bf(v);
  }
  __syncthreads();

  const int lane = tid & 31;
  const int wave = tid >> 5;
  const int hlf  = lane >> 4;
  const int m    = lane & 15;
  const int p0   = (wave & 3) * 32;   // 32 pixels per wave
  const int dr   = wave >> 2;         // which of the 2 output rows
  const int hh   = h0 + dr;

  // acc[g][i2]: pixel-group g (16 px each) x cout tile i2 (16 couts each)
  v8f acc[2][4];
  #pragma unroll
  for (int g = 0; g < 2; ++g)
    #pragma unroll
    for (int i2 = 0; i2 < 4; ++i2)
      #pragma unroll
      for (int j = 0; j < 8; ++j)
        acc[g][i2][j] = sBias[i2 * 16 + j + hlf * 8];

  #pragma unroll
  for (int t = 0; t < 9; ++t) {
    const int r = t / 3, s = t - 3 * r;
    #pragma unroll
    for (int q = 0; q < 2; ++q) {
      // A fragments (weights 16x32): lane<16 -> K 0..15, lane>=16 -> K 16..31
      const int chbA = q * 32 + hlf * 16;
      union { u32x4 u[2]; v16bf v; } aW[4];
      #pragma unroll
      for (int i2 = 0; i2 < 4; ++i2) {
        const uint16_t* pa = sW + (t * 64 + i2 * 16 + m) * 72 + chbA;
        aW[i2].u[0] = *(const u32x4*)(pa);
        aW[i2].u[1] = *(const u32x4*)(pa + 8);
      }
      // B fragments (input 32x16): lane<16 -> K 0..15, lane>=16 -> K 16..31
      const int chbB = q * 32 + hlf * 16;
      #pragma unroll
      for (int g = 0; g < 2; ++g) {
        const uint16_t* pb = sIn + (p0 + g * 16 + m + s) * 264 + (r + dr) * 64 + chbB;
        union { u32x4 u[2]; v16bf v; } bX;
        bX.u[0] = *(const u32x4*)(pb);
        bX.u[1] = *(const u32x4*)(pb + 8);
        #pragma unroll
        for (int i2 = 0; i2 < 4; ++i2)
          acc[g][i2] = __builtin_amdgcn_wmma_f32_16x16x32_bf16(
              false, aW[i2].v, false, bX.v, (short)0, acc[g][i2], false, false);
      }
    }
  }

  // ---- store: lane = pixel (coalesced along w)
  #pragma unroll
  for (int g = 0; g < 2; ++g) {
    const int w_out = p0 + g * 16 + m;
    if (w_out < W_) {
      #pragma unroll
      for (int i2 = 0; i2 < 4; ++i2)
        #pragma unroll
        for (int j = 0; j < 8; ++j) {
          int cout = i2 * 16 + j + hlf * 8;
          out[(((size_t)n * COUT_ + cout) * H_ + hh) * W_ + w_out] = acc[g][i2][j];
        }
    }
  }
}

extern "C" void kernel_launch(void* const* d_in, const int* in_sizes, int n_in,
                              void* d_out, int out_size, void* d_ws, size_t ws_size,
                              hipStream_t stream) {
  const float* x = (const float*)d_in[0];
  const float* w = (const float*)d_in[1];
  const float* b = (const float*)d_in[2];
  float* out  = (float*)d_out;
  uint16_t* wq = (uint16_t*)d_ws;   // 73,728 B of scratch for bf16 weights

  (void)hipFuncSetAttribute((const void*)conv_wmma,
                            hipFuncAttributeMaxDynamicSharedMemorySize,
                            SMEM_BYTES);

  prep_weights<<<(9 * 64 * 64 + 255) / 256, 256, 0, stream>>>(w, wq);
  conv_wmma<<<NB_ * (H_ / 2), 256, SMEM_BYTES, stream>>>(x, wq, b, out);
}
